// TCA_4312147165672
// MI455X (gfx1250) — matrix operand order repaired and checked by reading, not compiled, not run on hardware
//
#include <hip/hip_runtime.h>
#include <hip/hip_bf16.h>
#include <math.h>

// CDNA5 / gfx1250: wave32, WMMA 16x16x32 bf16 with f32 accumulate.
typedef __attribute__((ext_vector_type(16))) __bf16 v16bf;
typedef __attribute__((ext_vector_type(8)))  __bf16 v8bf;
typedef __attribute__((ext_vector_type(8)))  float  v8f;
typedef __attribute__((ext_vector_type(4)))  float  v4f;

#define B_ 32
#define L_ 1024
#define D_ 1024
#define F_ 512
#define C_ 256

// Branch-free tanh good to bf16 precision; correct saturation at +/-inf.
__device__ __forceinline__ float fast_tanh(float x) {
  return 1.0f - 2.0f / (__expf(2.0f * x) + 1.0f);
}

// ---------------------------------------------------------------------------
// K0: elementwise f32 -> bf16 (used for W_label and S_h; both tiny)
// ---------------------------------------------------------------------------
__global__ __launch_bounds__(256) void tca_cvt_bf16(
    const float* __restrict__ in, __bf16* __restrict__ outp, int n) {
  int i = blockIdx.x * 256 + threadIdx.x;
  if (i < n) outp[i] = (__bf16)in[i];
}

// ---------------------------------------------------------------------------
// K1: O[b,l,f] = tanh( sum_d X[b,l,d] * W[f,d] ), bf16 [B,L,F]
// grid (L/16, B); block 256 = 8 waves; wave -> 16(l) x 64(f) so one block
// covers full F=512 for one l-tile. Ping-pong stages s0/s1 reloaded in place
// (no cross-stage register copies), 1 A load feeds 4 WMMAs.
// ---------------------------------------------------------------------------
struct Stage1 {
  v4f  a[4];   // raw f32 A chunks: K {16h..16h+15}
  v16bf b[4];  // 4 B tiles (bf16)
};

__device__ __forceinline__ Stage1 load_stage1(
    const float* xrow, const __bf16* wr0, const __bf16* wr1,
    const __bf16* wr2, const __bf16* wr3, int k, int h) {
  Stage1 s;
  const float* p = xrow + k + h * 16;
  s.a[0] = *(const v4f*)(p);
  s.a[1] = *(const v4f*)(p + 4);
  s.a[2] = *(const v4f*)(p + 8);
  s.a[3] = *(const v4f*)(p + 12);
  s.b[0] = *(const v16bf*)(wr0 + k);
  s.b[1] = *(const v16bf*)(wr1 + k);
  s.b[2] = *(const v16bf*)(wr2 + k);
  s.b[3] = *(const v16bf*)(wr3 + k);
  return s;
}

__device__ __forceinline__ v16bf cvt_a16(const Stage1& s) {
  v16bf a;
#pragma unroll
  for (int j = 0; j < 4; ++j) {
    a[j]      = (__bf16)s.a[0][j];
    a[4 + j]  = (__bf16)s.a[1][j];
    a[8 + j]  = (__bf16)s.a[2][j];
    a[12 + j] = (__bf16)s.a[3][j];
  }
  return a;
}

__global__ __launch_bounds__(256) void tca_gemm1_tanh(
    const float* __restrict__ X,    // [B,L,D] f32
    const __bf16* __restrict__ Wb,  // [F,D]   bf16
    __bf16* __restrict__ O)         // [B,L,F] bf16
{
  const int lane = threadIdx.x & 31;
  const int wave = threadIdx.x >> 5;
  const int l0 = blockIdx.x * 16;
  const int b  = blockIdx.y;
  const int mn = lane & 15;  // A row / B column within tile
  const int h  = lane >> 4;  // lane-half -> K sub-range
  const int f0 = wave * 64;

  const float* xrow = X + ((size_t)b * L_ + (size_t)(l0 + mn)) * D_;
  const __bf16* wr0 = Wb + (size_t)(f0 +  0 + mn) * D_ + h * 16;
  const __bf16* wr1 = Wb + (size_t)(f0 + 16 + mn) * D_ + h * 16;
  const __bf16* wr2 = Wb + (size_t)(f0 + 32 + mn) * D_ + h * 16;
  const __bf16* wr3 = Wb + (size_t)(f0 + 48 + mn) * D_ + h * 16;

  v8f acc0 = {}, acc1 = {}, acc2 = {}, acc3 = {};

  auto do4 = [&](const Stage1& s) {
    v16bf a = cvt_a16(s);
    acc0 = __builtin_amdgcn_wmma_f32_16x16x32_bf16(false, a, false, s.b[0], (short)0, acc0, false, false);
    acc1 = __builtin_amdgcn_wmma_f32_16x16x32_bf16(false, a, false, s.b[1], (short)0, acc1, false, false);
    acc2 = __builtin_amdgcn_wmma_f32_16x16x32_bf16(false, a, false, s.b[2], (short)0, acc2, false, false);
    acc3 = __builtin_amdgcn_wmma_f32_16x16x32_bf16(false, a, false, s.b[3], (short)0, acc3, false, false);
  };

  Stage1 s0 = load_stage1(xrow, wr0, wr1, wr2, wr3, 0, h);
  Stage1 s1 = load_stage1(xrow, wr0, wr1, wr2, wr3, 32, h);
  for (int k = 0; k < D_ - 64; k += 64) {
    do4(s0);
    s0 = load_stage1(xrow, wr0, wr1, wr2, wr3, k + 64, h);
    do4(s1);
    s1 = load_stage1(xrow, wr0, wr1, wr2, wr3, k + 96, h);
  }
  do4(s0);
  do4(s1);

  // C/D layout: VGPR r, lane -> (M = r + 8h, N = lane&15)
  __bf16* orow = O + ((size_t)b * L_ + (size_t)l0) * F_ + f0 + mn;
#pragma unroll
  for (int r = 0; r < 8; ++r) {
    const size_t ro = (size_t)(r + 8 * h) * F_;
    orow[ro +  0] = (__bf16)fast_tanh(acc0[r]);
    orow[ro + 16] = (__bf16)fast_tanh(acc1[r]);
    orow[ro + 32] = (__bf16)fast_tanh(acc2[r]);
    orow[ro + 48] = (__bf16)fast_tanh(acc3[r]);
  }
}

// ---------------------------------------------------------------------------
// K2: scores[b,l,c] = sum_f O[b,l,f] * S[c,f], f32 [B,L,C], coalesced stores
// (lane dim = c). grid (L/16, B); wave -> 16(l) x 32(c); block covers C=256.
// Same ping-pong pipelining as K1.
// ---------------------------------------------------------------------------
struct Stage2 {
  v8bf a0, a1;   // A chunks (bf16)
  v16bf b[2];    // 2 B tiles
};

__device__ __forceinline__ Stage2 load_stage2(
    const __bf16* arow, const __bf16* sr0, const __bf16* sr1, int k, int h) {
  Stage2 s;
  s.a0 = *(const v8bf*)(arow + k + h * 16);
  s.a1 = *(const v8bf*)(arow + k + h * 16 + 8);
  s.b[0] = *(const v16bf*)(sr0 + k);
  s.b[1] = *(const v16bf*)(sr1 + k);
  return s;
}

__global__ __launch_bounds__(256) void tca_gemm2_scores(
    const __bf16* __restrict__ O,   // [B,L,F] bf16
    const __bf16* __restrict__ Sb,  // [C,F]   bf16
    float* __restrict__ scores)     // [B,L,C] f32
{
  const int lane = threadIdx.x & 31;
  const int wave = threadIdx.x >> 5;
  const int l0 = blockIdx.x * 16;
  const int b  = blockIdx.y;
  const int mn = lane & 15;
  const int h  = lane >> 4;
  const int c0 = wave * 32;

  const __bf16* arow = O + ((size_t)b * L_ + (size_t)(l0 + mn)) * F_;
  const __bf16* sr0 = Sb + (size_t)(c0 +  0 + mn) * F_ + h * 16;
  const __bf16* sr1 = Sb + (size_t)(c0 + 16 + mn) * F_ + h * 16;

  v8f acc0 = {}, acc1 = {};

  auto do2 = [&](const Stage2& s) {
    v16bf a = __builtin_shufflevector(s.a0, s.a1, 0, 1, 2, 3, 4, 5, 6, 7,
                                      8, 9, 10, 11, 12, 13, 14, 15);
    acc0 = __builtin_amdgcn_wmma_f32_16x16x32_bf16(false, a, false, s.b[0], (short)0, acc0, false, false);
    acc1 = __builtin_amdgcn_wmma_f32_16x16x32_bf16(false, a, false, s.b[1], (short)0, acc1, false, false);
  };

  Stage2 s0 = load_stage2(arow, sr0, sr1, 0, h);
  Stage2 s1 = load_stage2(arow, sr0, sr1, 32, h);
  for (int k = 0; k < F_ - 64; k += 64) {
    do2(s0);
    s0 = load_stage2(arow, sr0, sr1, k + 64, h);
    do2(s1);
    s1 = load_stage2(arow, sr0, sr1, k + 96, h);
  }
  do2(s0);
  do2(s1);

  // Store [B,L,C]: row l0+M, columns c0+N -> lanes 0..15 contiguous (64B).
  float* srow = scores + ((size_t)b * L_ + (size_t)l0) * C_ + c0 + mn;
#pragma unroll
  for (int r = 0; r < 8; ++r) {
    const size_t ro = (size_t)(r + 8 * h) * C_;
    srow[ro +  0] = acc0[r];
    srow[ro + 16] = acc1[r];
  }
}

// ---------------------------------------------------------------------------
// K3: partial softmax stats over l-chunks. grid (B, 4); block 256 threads,
// thread = c. Chunk q covers l in [256q, 256q+256). Coalesced: each l-step
// reads 256 consecutive c values. Two passes (reread hits L2).
// ---------------------------------------------------------------------------
__global__ __launch_bounds__(256) void tca_softmax_stats(
    const float* __restrict__ scores,  // [B,L,C]
    float* __restrict__ pm,            // [B,4,C] partial max
    float* __restrict__ pz)            // [B,4,C] partial sum
{
  const int b = blockIdx.x;
  const int q = blockIdx.y;
  const int c = threadIdx.x;
  const float* p = scores + ((size_t)b * L_ + (size_t)q * 256) * C_ + c;

  float m = -INFINITY;
  for (int l = 0; l < 256; ++l) m = fmaxf(m, p[(size_t)l * C_]);
  float z = 0.0f;
  for (int l = 0; l < 256; ++l) z += __expf(p[(size_t)l * C_] - m);

  const size_t o = ((size_t)b * 4 + q) * C_ + c;
  pm[o] = m;
  pz[o] = z;
}

// ---------------------------------------------------------------------------
// K4: combine partial stats -> m[b,c], invZ[b,c]. grid (B), block 256 (=C).
// ---------------------------------------------------------------------------
__global__ __launch_bounds__(256) void tca_softmax_combine(
    const float* __restrict__ pm, const float* __restrict__ pz,
    float* __restrict__ m_out, float* __restrict__ iz_out)
{
  const int b = blockIdx.x;
  const int c = threadIdx.x;
  const size_t base = (size_t)b * 4 * C_ + c;
  float m = pm[base];
  m = fmaxf(m, pm[base + C_]);
  m = fmaxf(m, pm[base + 2 * C_]);
  m = fmaxf(m, pm[base + 3 * C_]);
  float z = 0.0f;
#pragma unroll
  for (int q = 0; q < 4; ++q)
    z += pz[base + (size_t)q * C_] * __expf(pm[base + (size_t)q * C_] - m);
  m_out[(size_t)b * C_ + c] = m;
  iz_out[(size_t)b * C_ + c] = 1.0f / z;
}

// ---------------------------------------------------------------------------
// K5: w[b,l] = (1/C) * sum_c exp(s[b,l,c]-m[b,c]) * invZ[b,c].
// grid (B, L/256), block 256; thread = l, streams its contiguous 1KB row.
// ---------------------------------------------------------------------------
__global__ __launch_bounds__(256) void tca_make_w(
    const float* __restrict__ scores,  // [B,L,C]
    const float* __restrict__ m_in,    // [B,C]
    const float* __restrict__ iz_in,   // [B,C]
    float* __restrict__ w)             // [B,L]
{
  __shared__ float sm[C_];
  __shared__ float siz[C_];
  const int b = blockIdx.x;
  const int l = blockIdx.y * 256 + threadIdx.x;
  sm[threadIdx.x]  = m_in[(size_t)b * C_ + threadIdx.x];
  siz[threadIdx.x] = iz_in[(size_t)b * C_ + threadIdx.x];
  __syncthreads();

  const float* row = scores + ((size_t)b * L_ + l) * C_;
  float acc = 0.0f;
  for (int c = 0; c < C_; c += 4) {
    v4f s = *(const v4f*)(row + c);
    acc += __expf(s.x - sm[c + 0]) * siz[c + 0];
    acc += __expf(s.y - sm[c + 1]) * siz[c + 1];
    acc += __expf(s.z - sm[c + 2]) * siz[c + 2];
    acc += __expf(s.w - sm[c + 3]) * siz[c + 3];
  }
  w[(size_t)b * L_ + l] = acc * (1.0f / (float)C_);
}

// ---------------------------------------------------------------------------
// K6: out[b,d] = sum_l w[b,l] * X[b,l,d].  grid (B, D/256), block 256.
// ---------------------------------------------------------------------------
__global__ __launch_bounds__(256) void tca_weighted_sum(
    const float* __restrict__ X,  // [B,L,D]
    const float* __restrict__ w,  // [B,L]
    float* __restrict__ out)      // [B,D]
{
  __shared__ float wl[L_];
  const int b = blockIdx.x;
  const int d = blockIdx.y * 256 + threadIdx.x;
  for (int i = threadIdx.x; i < L_; i += 256) wl[i] = w[(size_t)b * L_ + i];
  __syncthreads();
  const float* xp = X + (size_t)b * L_ * D_ + d;
  float acc = 0.0f;
  for (int l = 0; l < L_; ++l) acc += wl[l] * xp[(size_t)l * D_];
  out[(size_t)b * D_ + d] = acc;
}

// ---------------------------------------------------------------------------
extern "C" void kernel_launch(void* const* d_in, const int* in_sizes, int n_in,
                              void* d_out, int out_size, void* d_ws, size_t ws_size,
                              hipStream_t stream) {
  const float* X = (const float*)d_in[0];  // word_x  [B,L,D]
  const float* W = (const float*)d_in[1];  // W_label [F,D]
  const float* S = (const float*)d_in[2];  // S_h     [C,F]
  float* out = (float*)d_out;              // [B,D]

  // Workspace layout (all offsets 256B-aligned)
  char* ws = (char*)d_ws;
  size_t off = 0;
  __bf16* O = (__bf16*)(ws + off);      off += (size_t)B_ * L_ * F_ * 2;  // 32 MiB
  float* scores = (float*)(ws + off);   off += (size_t)B_ * L_ * C_ * 4;  // 32 MiB
  __bf16* Wb = (__bf16*)(ws + off);     off += (size_t)F_ * D_ * 2;
  __bf16* Sb = (__bf16*)(ws + off);     off += (size_t)C_ * F_ * 2;
  float* pm = (float*)(ws + off);       off += (size_t)B_ * 4 * C_ * 4;
  float* pz = (float*)(ws + off);       off += (size_t)B_ * 4 * C_ * 4;
  float* mbc = (float*)(ws + off);      off += (size_t)B_ * C_ * 4;
  float* izbc = (float*)(ws + off);     off += (size_t)B_ * C_ * 4;
  float* wvec = (float*)(ws + off);     off += (size_t)B_ * L_ * 4;

  tca_cvt_bf16<<<(F_ * D_) / 256, 256, 0, stream>>>(W, Wb, F_ * D_);
  tca_cvt_bf16<<<(C_ * F_) / 256, 256, 0, stream>>>(S, Sb, C_ * F_);
  tca_gemm1_tanh<<<dim3(L_ / 16, B_), 256, 0, stream>>>(X, Wb, O);
  tca_gemm2_scores<<<dim3(L_ / 16, B_), 256, 0, stream>>>(O, Sb, scores);
  tca_softmax_stats<<<dim3(B_, 4), 256, 0, stream>>>(scores, pm, pz);
  tca_softmax_combine<<<B_, 256, 0, stream>>>(pm, pz, mbc, izbc);
  tca_make_w<<<dim3(B_, L_ / 256), 256, 0, stream>>>(scores, mbc, izbc, wvec);
  tca_weighted_sum<<<dim3(B_, D_ / 256), 256, 0, stream>>>(X, wvec, out);
}
